// SomLayer_26517128086090
// MI455X (gfx1250) — matrix operand order repaired and checked by reading, not compiled, not run on hardware
//
#include <hip/hip_runtime.h>

// ---------------------------------------------------------------------------
// SOM / VQ layer for MI455X (gfx1250, wave32, WMMA).
//   x: [65536, 64] f32, W: [4096, 64] f32
//   out: quantized [65536,64] f32  ++  diff scalar (1 float)
//
// v2: LDS-staged codebook. Each 8-wave block shares a 128-code chunk of the
// bf16 hi/lo split codebook in LDS (37 KB, bank-conflict-padded rows), with a
// software pipeline: next chunk is prefetched global->VGPR while the current
// chunk's 192 WMMAs run, then ds_store + barrier. Inner loop has no VMEM.
// Cuts codebook L2 traffic 8x (1 GB -> 128 MB) and removes the per-k-tile
// s_wait_loadcnt stall seen in v1.
// ---------------------------------------------------------------------------

typedef __attribute__((ext_vector_type(16))) __bf16 v16bf;
typedef __attribute__((ext_vector_type(8)))  __bf16 v8bf;
typedef __attribute__((ext_vector_type(8)))  float  v8f;
typedef __attribute__((ext_vector_type(4)))  float  v4f;

#define N_ROWS  65536
#define K_CODES 4096
#define DIM     64
#define NELEM   (N_ROWS * DIM)       // 4194304
#define MT      4                     // 16-row M-tiles per wave (64 rows/wave)
#define NWAVES  (N_ROWS / (MT * 16))  // 1024 waves

#define CHUNK   128                   // codes staged in LDS at a time
#define LSTRIDE 72                    // padded LDS row (bf16): 144B = 36 banks
#define NCHUNK  (K_CODES / CHUNK)     // 32
#define KTPC    (CHUNK / 16)          // 8 k-tiles per chunk

static __device__ __forceinline__ v8f wmma_bf16(v16bf a, v16bf b, v8f c) {
  return __builtin_amdgcn_wmma_f32_16x16x32_bf16(
      /*neg_a=*/false, a, /*neg_b=*/false, b,
      /*c_mod=*/(short)0, c, /*reuse_a=*/false, /*reuse_b=*/false);
}

// 32B (16 x bf16) load as two 16B pieces (16B alignment guaranteed by layout)
static __device__ __forceinline__ v16bf ld32(const __bf16* p) {
  union { v16bf v; v8bf h[2]; } u;
  u.h[0] = *(const v8bf*)p;
  u.h[1] = *(const v8bf*)(p + 8);
  return u.v;
}

// --- prep: per-code squared norms + bf16 hi/lo split of the codebook -------
__global__ void som_prep(const float* __restrict__ W,
                         float* __restrict__ wnorm,
                         __bf16* __restrict__ Whi,
                         __bf16* __restrict__ Wlo) {
  int k = blockIdx.x * blockDim.x + threadIdx.x;
  if (k >= K_CODES) return;
  float acc = 0.f;
  #pragma unroll 8
  for (int j = 0; j < DIM; ++j) {
    float w = W[k * DIM + j];
    acc = __builtin_fmaf(w, w, acc);
    __bf16 h = (__bf16)w;
    Whi[k * DIM + j] = h;
    Wlo[k * DIM + j] = (__bf16)(w - (float)h);
  }
  wnorm[k] = acc;
}

// --- main: per-wave 64 rows; block shares LDS-staged codebook chunks -------
__global__ __launch_bounds__(256)
void som_wmma(const float* __restrict__ x,
              const float* __restrict__ W,
              const float* __restrict__ wnorm,
              const __bf16* __restrict__ Whi,
              const __bf16* __restrict__ Wlo,
              float* __restrict__ out,
              float* __restrict__ wavesum) {
  __shared__ __bf16 sh_hi[CHUNK * LSTRIDE];   // 18 KB
  __shared__ __bf16 sh_lo[CHUNK * LSTRIDE];   // 18 KB
  __shared__ float  sh_wn[CHUNK];             // 512 B

  const int t       = threadIdx.x;
  const int lane    = t & 31;
  const int wave    = (blockIdx.x * blockDim.x + t) >> 5;
  const int l16     = lane & 15;
  const int g       = lane >> 4;              // half-wave group
  const int rowBase = wave * (MT * 16);

  // ---- Load x rows, split into bf16 hi/lo as WMMA A operands.
  // A 16x32 bf16, contiguous K order used here (B matches): lanes 0-15 -> M=lane, K in {0..15};
  //               lanes 16-31 -> K in {16..31}.  Two K-chunks (c).
  v16bf ah[MT][2], al[MT][2];
  #pragma unroll
  for (int mt = 0; mt < MT; ++mt) {
    const float* xr = x + (size_t)(rowBase + mt * 16 + l16) * DIM;
    #pragma unroll
    for (int c = 0; c < 2; ++c) {
      const int k0 = c * 32 + g * 16;
      #pragma unroll
      for (int e = 0; e < 8; ++e) {
        float v0 = xr[k0 + e];
        float v1 = xr[k0 + 8 + e];
        __bf16 h0 = (__bf16)v0;
        __bf16 h1 = (__bf16)v1;
        ah[mt][c][e]     = h0;
        ah[mt][c][e + 8] = h1;
        al[mt][c][e]     = (__bf16)(v0 - (float)h0);
        al[mt][c][e + 8] = (__bf16)(v1 - (float)h1);
      }
    }
  }

  // ---- Pipeline staging registers (64B of hi + 64B of lo per thread).
  uint4 rh[4], rl[4];
  float rwn = 0.f;
  const int srow = t >> 3;            // 0..31 (row within i-th 32-row slab)
  const int scol = (t & 7) * 8;       // bf16 column offset, 16B granules

  auto fetch = [&](int ci) {          // global -> VGPR (coalesced 4KB/instr)
    const __bf16* gh = Whi + (size_t)ci * (CHUNK * DIM);
    const __bf16* gl = Wlo + (size_t)ci * (CHUNK * DIM);
    #pragma unroll
    for (int i = 0; i < 4; ++i) {
      rh[i] = *(const uint4*)(gh + i * 2048 + t * 8);
      rl[i] = *(const uint4*)(gl + i * 2048 + t * 8);
    }
    if (t < CHUNK) rwn = wnorm[ci * CHUNK + t];
  };
  auto stage = [&]() {                // VGPR -> LDS (padded rows)
    #pragma unroll
    for (int i = 0; i < 4; ++i) {
      *(uint4*)(sh_hi + (i * 32 + srow) * LSTRIDE + scol) = rh[i];
      *(uint4*)(sh_lo + (i * 32 + srow) * LSTRIDE + scol) = rl[i];
    }
    if (t < CHUNK) sh_wn[t] = rwn;
  };

  // ---- Running argmin state: lane covers column n = code%16 == l16; C-layout
  // VGPR r is row (g*8 + r) of the current M-tile.
  float bs[MT][8];
  int   bi[MT][8];
  #pragma unroll
  for (int mt = 0; mt < MT; ++mt)
    #pragma unroll
    for (int r = 0; r < 8; ++r) { bs[mt][r] = 3.4e38f; bi[mt][r] = 0; }

  fetch(0);
  stage();
  __syncthreads();

  for (int ci = 0; ci < NCHUNK; ++ci) {
    if (ci + 1 < NCHUNK) fetch(ci + 1);   // latency hidden under 192 WMMAs

    #pragma unroll 2
    for (int ktl = 0; ktl < KTPC; ++ktl) {
      const int local = ktl * 16 + l16;
      const int code  = ci * CHUNK + local;
      // B 32x16 bf16: lanes 0-15 hold K=0..15, lanes 16-31 hold K=16..31.
      const __bf16* ph = sh_hi + local * LSTRIDE + g * 16;
      const __bf16* pl = sh_lo + local * LSTRIDE + g * 16;
      v16bf bh0 = ld32(ph);
      v16bf bh1 = ld32(ph + 32);
      v16bf bl0 = ld32(pl);
      v16bf bl1 = ld32(pl + 32);
      float wn  = sh_wn[local];

      #pragma unroll
      for (int mt = 0; mt < MT; ++mt) {
        v8f acc = {};
        // dot ~= xh*wh + xl*wh + xh*wl  (fp32 accumulate), both 32-K chunks
        acc = wmma_bf16(ah[mt][0], bh0, acc);
        acc = wmma_bf16(al[mt][0], bh0, acc);
        acc = wmma_bf16(ah[mt][0], bl0, acc);
        acc = wmma_bf16(ah[mt][1], bh1, acc);
        acc = wmma_bf16(al[mt][1], bh1, acc);
        acc = wmma_bf16(ah[mt][1], bl1, acc);
        #pragma unroll
        for (int r = 0; r < 8; ++r) {
          float s = __builtin_fmaf(-2.f, acc[r], wn);  // ||w||^2 - 2 x.w
          if (s < bs[mt][r]) { bs[mt][r] = s; bi[mt][r] = code; }
        }
      }
    }

    __syncthreads();                       // everyone done reading this chunk
    if (ci + 1 < NCHUNK) {
      stage();                             // overwrite LDS with next chunk
      __syncthreads();
    }
  }

  // ---- Cross-lane argmin: butterfly within each 16-lane half (xor masks <16
  // never cross bit 4, so rows 0-7 reduce in lanes 0-15, rows 8-15 in 16-31).
  #pragma unroll
  for (int mt = 0; mt < MT; ++mt)
    #pragma unroll
    for (int r = 0; r < 8; ++r) {
      float s = bs[mt][r];
      int   i = bi[mt][r];
      #pragma unroll
      for (int off = 8; off >= 1; off >>= 1) {
        float os = __shfl_xor(s, off, 32);
        int   oi = __shfl_xor(i, off, 32);
        if (os < s || (os == s && oi < i)) { s = os; i = oi; }
      }
      bi[mt][r] = i;  // bmu for row (mt*16 + g*8 + r), same in all 16 lanes
    }

  // ---- Gather W[bmu] -> out, accumulate per-wave sum of (q-x)^2.
  float dsum = 0.f;
  #pragma unroll
  for (int mt = 0; mt < MT; ++mt) {
    #pragma unroll
    for (int r = 0; r < 8; ++r) {
      const int row  = rowBase + mt * 16 + g * 8 + r;
      const int code = bi[mt][r];
      v4f wv = *(const v4f*)(W + (size_t)code * DIM + l16 * 4);
      v4f xv = *(const v4f*)(x + (size_t)row  * DIM + l16 * 4);
      *(v4f*)(out + (size_t)row * DIM + l16 * 4) = wv;
      #pragma unroll
      for (int j = 0; j < 4; ++j) {
        float d = wv[j] - xv[j];
        dsum = __builtin_fmaf(d, d, dsum);
      }
    }
  }
  #pragma unroll
  for (int off = 16; off >= 1; off >>= 1)
    dsum += __shfl_xor(dsum, off, 32);
  if (lane == 0) wavesum[wave] = dsum;     // deterministic: one slot per wave
}

// --- fixed-tree reduction of the 1024 per-wave partials -> mean ------------
__global__ void som_diff_reduce(const float* __restrict__ wavesum,
                                float* __restrict__ diff) {
  int lane = threadIdx.x;                  // launched with 32 threads
  float s = 0.f;
  for (int i = lane; i < NWAVES; i += 32) s += wavesum[i];
  #pragma unroll
  for (int off = 16; off >= 1; off >>= 1)
    s += __shfl_xor(s, off, 32);
  if (lane == 0) *diff = s * (1.0f / (float)NELEM);
}

extern "C" void kernel_launch(void* const* d_in, const int* in_sizes, int n_in,
                              void* d_out, int out_size, void* d_ws, size_t ws_size,
                              hipStream_t stream) {
  const float* x = (const float*)d_in[0];    // [16,4096,64] f32
  const float* W = (const float*)d_in[1];    // [4096,64] f32
  float* out  = (float*)d_out;               // quantized ++ diff
  float* diff = out + NELEM;

  // workspace layout: wnorm(16KB) | Whi(512KB) | Wlo(512KB) | wavesum(4KB)
  float*  wnorm   = (float*)d_ws;
  __bf16* Whi     = (__bf16*)((char*)d_ws + 16 * 1024);
  __bf16* Wlo     = Whi + (size_t)K_CODES * DIM;
  float*  wavesum = (float*)((char*)d_ws + 16 * 1024 + 2 * 512 * 1024);

  som_prep<<<K_CODES / 256, 256, 0, stream>>>(W, wnorm, Whi, Wlo);
  som_wmma<<<(NWAVES * 32) / 256, 256, 0, stream>>>(x, W, wnorm, Whi, Wlo,
                                                    out, wavesum);
  som_diff_reduce<<<1, 32, 0, stream>>>(wavesum, diff);
}
